// MultiHeadDilatedAttention_43894565765420
// MI455X (gfx1250) — matrix operand order repaired and checked by reading, not compiled, not run on hardware
//
#include <hip/hip_runtime.h>
#include <cstddef>

// ---------------------------------------------------------------------------
// MultiHeadDilatedAttention for MI455X (gfx1250, wave32, WMMA).
//
// Key optimization: dilation makes the attention *diagonal* and head h only
// uses positions p with p%4 == h%4. We therefore compute in a compact layout
// [bc = b*4 + (p%4)][i = p/4][h4][d] of width 256 (4 heads x 64), cutting all
// four GEMMs' FLOPs by 4x. bf16 WMMA (v_wmma_f32_16x16x32_bf16), f32 accum.
// ---------------------------------------------------------------------------

typedef __attribute__((ext_vector_type(16))) __bf16 v16bf;
typedef __attribute__((ext_vector_type(8)))  __bf16 bf16x8;
typedef __attribute__((ext_vector_type(8)))  float  v8f;

#define D_MODEL   1024
#define HEAD_DIM  64
#define B_SZ      4
#define S_SZ      4096
#define DIL       4
#define SEG       512
#define SD        (SEG / DIL)        // 128 dilated positions per segment
#define NSEG      (S_SZ / SEG)       // 8
#define CPOS      (S_SZ / DIL)       // 1024 compact rows per (b,c)
#define CW        (4 * HEAD_DIM)     // 256 compact channels

static __device__ inline v16bf make_frag(const __bf16* p0, const __bf16* p1) {
  bf16x8 lo = *(const bf16x8*)p0;
  bf16x8 hi = *(const bf16x8*)p1;
  v16bf r;
#pragma unroll
  for (int i = 0; i < 8; ++i) { r[i] = lo[i]; r[8 + i] = hi[i]; }
  return r;
}

// ---------------------------------------------------------------------------
// Kernel 1: fused compact QKV projection.
// Per block: one (b,c) pair, one 128-row M tile, one 64-col head block (h4),
// computing Q,K,V simultaneously (A tile in LDS shared across 3 B matrices).
// grid = 16 (bc) * 8 (mt) * 4 (nt) = 512 blocks, 256 threads (8 waves).
// ---------------------------------------------------------------------------
__global__ __launch_bounds__(256)
void qkv_gemm_kernel(const float* __restrict__ x,
                     const float* __restrict__ Wq, const float* __restrict__ bq,
                     const float* __restrict__ Wk, const float* __restrict__ bk,
                     const float* __restrict__ Wv, const float* __restrict__ bv,
                     unsigned short* __restrict__ Qc_, unsigned short* __restrict__ Kc_,
                     unsigned short* __restrict__ Vc_) {
  __bf16* Qc = (__bf16*)Qc_;
  __bf16* Kc = (__bf16*)Kc_;
  __bf16* Vc = (__bf16*)Vc_;

  __shared__ __align__(16) __bf16 As[128][40];        // [m][k], pad stride 40
  __shared__ __align__(16) __bf16 BsT[3][64][40];     // [proj][n][k] transposed

  const int tid   = threadIdx.x;
  const int wid   = tid >> 5;
  const int lane  = tid & 31;
  const int laneR = lane & 15;
  const int half  = lane >> 4;

  int tmp = blockIdx.x;
  const int nt = tmp & 3;            // h4 (head-within-class)
  const int mt = (tmp >> 2) & 7;     // M tile
  const int bc = tmp >> 5;           // b*4 + c
  const int b  = bc >> 2;
  const int c  = bc & 3;

  const int colBase = (4 * nt + c) * HEAD_DIM;   // contiguous 64 columns of W
  const float* Ws[3] = { Wq, Wk, Wv };

  v8f acc[3][4];
#pragma unroll
  for (int w3 = 0; w3 < 3; ++w3)
#pragma unroll
    for (int s = 0; s < 4; ++s)
      acc[w3][s] = (v8f){0.f,0.f,0.f,0.f,0.f,0.f,0.f,0.f};

  for (int k0 = 0; k0 < D_MODEL; k0 += 32) {
    // ---- stage A: 128 rows (class-c positions, stride 4 in x) x 32 k, fp32->bf16
#pragma unroll
    for (int pass = 0; pass < 4; ++pass) {
      const int row = pass * 32 + (tid >> 3);
      const int c4  = (tid & 7) * 4;
      const int p   = 4 * (mt * 128 + row) + c;
      const float4 v = *(const float4*)&x[((size_t)b * S_SZ + p) * D_MODEL + k0 + c4];
      As[row][c4 + 0] = (__bf16)v.x;
      As[row][c4 + 1] = (__bf16)v.y;
      As[row][c4 + 2] = (__bf16)v.z;
      As[row][c4 + 3] = (__bf16)v.w;
      if (k0 + 32 < D_MODEL)
        __builtin_prefetch(&x[((size_t)b * S_SZ + p) * D_MODEL + k0 + 32 + c4], 0, 1);
    }
    // ---- stage B (transposed): 3 x (32 k x 64 n), fp32->bf16
#pragma unroll
    for (int w3 = 0; w3 < 3; ++w3) {
#pragma unroll
      for (int pass = 0; pass < 2; ++pass) {
        const int kk = pass * 16 + (tid >> 4);
        const int c4 = (tid & 15) * 4;
        const float4 v = *(const float4*)&Ws[w3][(size_t)(k0 + kk) * D_MODEL + colBase + c4];
        BsT[w3][c4 + 0][kk] = (__bf16)v.x;
        BsT[w3][c4 + 1][kk] = (__bf16)v.y;
        BsT[w3][c4 + 2][kk] = (__bf16)v.z;
        BsT[w3][c4 + 3][kk] = (__bf16)v.w;
      }
    }
    __syncthreads();

    // A fragment: lane layout per ISA 16-bit A 16x32 (K 0..7/16..23 | 8..15/24..31)
    const int mRow = wid * 16 + laneR;
    const v16bf a = make_frag(&As[mRow][half * 8], &As[mRow][16 + half * 8]);

#pragma unroll
    for (int w3 = 0; w3 < 3; ++w3) {
#pragma unroll
      for (int s = 0; s < 4; ++s) {
        const int col = s * 16 + laneR;
        const v16bf bb = make_frag(&BsT[w3][col][half * 8], &BsT[w3][col][16 + half * 8]);
        acc[w3][s] = __builtin_amdgcn_wmma_f32_16x16x32_bf16(
            false, a, false, bb, (short)0, acc[w3][s], false, false);
      }
    }
    __syncthreads();
  }

  // ---- epilogue: bias + bf16 store into compact Q/K/V
  const float* bias[3] = { bq, bk, bv };
  __bf16* dst[3] = { Qc, Kc, Vc };
#pragma unroll
  for (int w3 = 0; w3 < 3; ++w3) {
#pragma unroll
    for (int s = 0; s < 4; ++s) {
      const int jj = s * 16 + laneR;
      const float bia = bias[w3][colBase + jj];
#pragma unroll
      for (int vj = 0; vj < 8; ++vj) {
        const int m = half * 8 + vj;   // C/D layout: lanes 16-31 -> M = 8 + vgpr
        const size_t row = (size_t)bc * CPOS + mt * 128 + wid * 16 + m;
        dst[w3][row * CW + nt * HEAD_DIM + jj] = (__bf16)(acc[w3][s][vj] + bia);
      }
    }
  }
}

// ---------------------------------------------------------------------------
// Kernel 2: diagonal dilated attention. score_t = q_t . k_t / 8, softmax over
// the 128 dilated positions of one segment, scale V. Fully dense in compact
// layout (the zero-scatter of the reference is implicit).
// grid = 16 (bc) * 8 (n) * 4 (h4) = 512 blocks, 128 threads (t = position).
// ---------------------------------------------------------------------------
__global__ __launch_bounds__(128)
void dilated_attn_kernel(const unsigned short* __restrict__ Qc_,
                         const unsigned short* __restrict__ Kc_,
                         const unsigned short* __restrict__ Vc_,
                         unsigned short* __restrict__ Ac_) {
  const __bf16* Qc = (const __bf16*)Qc_;
  const __bf16* Kc = (const __bf16*)Kc_;
  const __bf16* Vc = (const __bf16*)Vc_;
  __bf16* Ac = (__bf16*)Ac_;

  __shared__ float sc[SD];
  const int tid = threadIdx.x;
  int tmp = blockIdx.x;
  const int h4 = tmp & 3;
  const int n  = (tmp >> 2) & 7;
  const int bc = tmp >> 5;

  const size_t row  = (size_t)bc * CPOS + n * SD + tid;
  const size_t base = row * CW + h4 * HEAD_DIM;

  float dot = 0.f;
#pragma unroll
  for (int ch = 0; ch < 8; ++ch) {
    const bf16x8 q = *(const bf16x8*)&Qc[base + ch * 8];
    const bf16x8 k = *(const bf16x8*)&Kc[base + ch * 8];
#pragma unroll
    for (int j = 0; j < 8; ++j) dot += (float)q[j] * (float)k[j];
  }
  const float score = dot * 0.125f;   // head_dim^-0.5 = 1/8
  sc[tid] = score;
  __syncthreads();

  float mx = -1e30f;
  for (int j = 0; j < SD; ++j) mx = fmaxf(mx, sc[j]);
  float den = 0.f;
  for (int j = 0; j < SD; ++j) den += __expf(sc[j] - mx);
  const float p = __expf(score - mx) / den;

#pragma unroll
  for (int ch = 0; ch < 8; ++ch) {
    const bf16x8 v = *(const bf16x8*)&Vc[base + ch * 8];
    bf16x8 o;
#pragma unroll
    for (int j = 0; j < 8; ++j) o[j] = (__bf16)(p * (float)v[j]);
    *(bf16x8*)&Ac[base + ch * 8] = o;
  }
}

// ---------------------------------------------------------------------------
// Kernel 3: output projection. out[b, 4*i + c, :] = Ac[bc, i, :] (256 wide)
// @ Wo[selected 256 rows for class c] + bo. K = 256 only (4x FLOP cut).
// grid = 16 (bc) * 8 (mt) * 8 (ntile) = 1024 blocks, 256 threads (8 waves).
// ---------------------------------------------------------------------------
__global__ __launch_bounds__(256)
void out_gemm_kernel(const unsigned short* __restrict__ Ac_,
                     const float* __restrict__ Wo, const float* __restrict__ bo,
                     float* __restrict__ out) {
  const __bf16* Ac = (const __bf16*)Ac_;

  __shared__ __align__(16) __bf16 As[128][40];    // [m][k]
  __shared__ __align__(16) __bf16 BsT[128][40];   // [n][k] transposed

  const int tid   = threadIdx.x;
  const int wid   = tid >> 5;
  const int lane  = tid & 31;
  const int laneR = lane & 15;
  const int half  = lane >> 4;

  int tmp = blockIdx.x;
  const int ntile = tmp & 7;
  const int mt    = (tmp >> 3) & 7;
  const int bc    = tmp >> 6;
  const int b     = bc >> 2;
  const int c     = bc & 3;
  const int nBase = ntile * 128;

  v8f acc[8];
#pragma unroll
  for (int s = 0; s < 8; ++s) acc[s] = (v8f){0.f,0.f,0.f,0.f,0.f,0.f,0.f,0.f};

  for (int k0 = 0; k0 < CW; k0 += 32) {   // 8 iterations
    // ---- stage A: 128 x 32 bf16 from compact attention
#pragma unroll
    for (int pass = 0; pass < 2; ++pass) {
      const int rr = pass * 64 + (tid >> 2);
      const int c8 = (tid & 3) * 8;
      const bf16x8 v =
          *(const bf16x8*)&Ac[((size_t)bc * CPOS + mt * 128 + rr) * CW + k0 + c8];
#pragma unroll
      for (int j = 0; j < 8; ++j) As[rr][c8 + j] = v[j];
    }
    // ---- stage B (transposed): 32 k x 128 n; map compact k -> Wo row for class c
#pragma unroll
    for (int pass = 0; pass < 4; ++pass) {
      const int kk = pass * 8 + (tid >> 5);
      const int c4 = (tid & 31) * 4;
      const int k  = k0 + kk;
      const int origRow = ((k >> 6) * 4 + c) * HEAD_DIM + (k & 63);
      const float4 v = *(const float4*)&Wo[(size_t)origRow * D_MODEL + nBase + c4];
      BsT[c4 + 0][kk] = (__bf16)v.x;
      BsT[c4 + 1][kk] = (__bf16)v.y;
      BsT[c4 + 2][kk] = (__bf16)v.z;
      BsT[c4 + 3][kk] = (__bf16)v.w;
    }
    __syncthreads();

    const int mRow = wid * 16 + laneR;
    const v16bf a = make_frag(&As[mRow][half * 8], &As[mRow][16 + half * 8]);
#pragma unroll
    for (int s = 0; s < 8; ++s) {
      const int col = s * 16 + laneR;
      const v16bf bb = make_frag(&BsT[col][half * 8], &BsT[col][16 + half * 8]);
      acc[s] = __builtin_amdgcn_wmma_f32_16x16x32_bf16(
          false, a, false, bb, (short)0, acc[s], false, false);
    }
    __syncthreads();
  }

  // ---- epilogue: scatter compact row i -> position p = 4*i + c, add bias
#pragma unroll
  for (int s = 0; s < 8; ++s) {
    const int col = nBase + s * 16 + laneR;
    const float bia = bo[col];
#pragma unroll
    for (int vj = 0; vj < 8; ++vj) {
      const int i = mt * 128 + wid * 16 + half * 8 + vj;
      const int p = 4 * i + c;
      out[((size_t)b * S_SZ + p) * D_MODEL + col] = acc[s][vj] + bia;
    }
  }
}

// ---------------------------------------------------------------------------
extern "C" void kernel_launch(void* const* d_in, const int* in_sizes, int n_in,
                              void* d_out, int out_size, void* d_ws, size_t ws_size,
                              hipStream_t stream) {
  (void)in_sizes; (void)n_in; (void)out_size; (void)ws_size;
  const float* x  = (const float*)d_in[0];
  const float* Wq = (const float*)d_in[1];
  const float* bq = (const float*)d_in[2];
  const float* Wk = (const float*)d_in[3];
  const float* bk = (const float*)d_in[4];
  const float* Wv = (const float*)d_in[5];
  const float* bv = (const float*)d_in[6];
  const float* Wo = (const float*)d_in[7];
  const float* bo = (const float*)d_in[8];
  float* out = (float*)d_out;

  // Workspace: 4 compact bf16 tensors of 16 * 1024 * 256 elements (8 MB each).
  const size_t cElems = (size_t)B_SZ * DIL * CPOS * CW;   // 4,194,304
  unsigned short* Qc = (unsigned short*)d_ws;
  unsigned short* Kc = Qc + cElems;
  unsigned short* Vc = Kc + cElems;
  unsigned short* Ac = Vc + cElems;

  qkv_gemm_kernel<<<512, 256, 0, stream>>>(x, Wq, bq, Wk, bk, Wv, bv, Qc, Kc, Vc);
  dilated_attn_kernel<<<512, 128, 0, stream>>>(Qc, Kc, Vc, Ac);
  out_gemm_kernel<<<1024, 256, 0, stream>>>(Ac, Wo, bo, out);
}
